// EmbeddingAndSeqformer_13675175870793
// MI455X (gfx1250) — matrix operand order repaired and checked by reading, not compiled, not run on hardware
//
#include <hip/hip_runtime.h>
#include <math.h>

// ---------------------------------------------------------------------------
// CDNA5 (gfx1250) wave32 WMMA types
// ---------------------------------------------------------------------------
typedef __attribute__((ext_vector_type(16))) _Float16 v16h;
typedef __attribute__((ext_vector_type(8)))  float    v8f;

// ---------------------------------------------------------------------------
// Generic batched GEMM, f32 in/out, f16 WMMA compute (16x16x32), f32 accum.
//   C = act(alpha * A x B + bias)
//   TA=0: A[m,k] (row stride lda)   TA=1: A[k,m] (row stride lda)
//   TB=0: B[k,n]                    TB=1: B[n,k]
// Two-level batching: z -> (zo = z / batch_inner, zi = z % batch_inner)
// Double-buffered LDS, software-pipelined staging (global_load_b128 fast path).
// ---------------------------------------------------------------------------
struct GP {
  const float* A; const float* B; const float* bias; float* C;
  int M, N, K;
  int lda, ldb, ldc;
  int act;            // 0 none, 1 relu, 2 sigmoid
  int batch_inner;
  float alpha;
  long long sAi, sAo, sBi, sBo, sCi, sCo;
};

template <int TA, int TB>
__global__ __launch_bounds__(256) void k_gemm(GP p) {
  __shared__ _Float16 As[2][64][36];    // [buf][m][k], padded
  __shared__ _Float16 Bs[2][128][36];   // [buf][n][k], padded (B transposed in LDS)

  const int tid = threadIdx.x;
  const int zi = (int)(blockIdx.z % (unsigned)p.batch_inner);
  const int zo = (int)(blockIdx.z / (unsigned)p.batch_inner);
  const float* Ab = p.A + (long long)zi * p.sAi + (long long)zo * p.sAo;
  const float* Bb = p.B + (long long)zi * p.sBi + (long long)zo * p.sBo;
  float* Cb = p.C + (long long)zi * p.sCi + (long long)zo * p.sCo;

  const int mblk = blockIdx.y * 64;
  const int nblk = blockIdx.x * 128;

  const int wave  = tid >> 5;
  const int wm    = (wave >> 2) * 32;   // 0 or 32
  const int wn    = (wave & 3) * 32;    // 0..96
  const int lane  = tid & 31;
  const int lmod  = lane & 15;
  const int lhalf = lane >> 4;

  const bool fastAm = (mblk + 64 <= p.M);
  const bool fastBn = (nblk + 128 <= p.N);

  float4 ar[2];   // staging regs: A tile 64x32 = 512 float4 / 256 threads
  float4 br[4];   // staging regs: B tile 128x32 = 1024 float4 / 256 threads

  // ---- issue global loads for one K-stage into registers ----
  auto gload = [&](int k0) {
    const bool fk = (k0 + 32 <= p.K);
    if (TA == 0) {
      if (fastAm && fk) {
#pragma unroll
        for (int i = 0; i < 2; ++i) {
          int idx = tid + i * 256;
          int m = idx >> 3, kq = (idx & 7) * 4;
          ar[i] = *(const float4*)(Ab + (long long)(mblk + m) * p.lda + (k0 + kq));
        }
      } else {
#pragma unroll
        for (int i = 0; i < 2; ++i) {
          int idx = tid + i * 256;
          int m = idx >> 3, kq = (idx & 7) * 4;
          int gm = mblk + m;
          int cm = gm < p.M ? gm : p.M - 1;
          float* o = (float*)&ar[i];
#pragma unroll
          for (int u = 0; u < 4; ++u) {
            int gk = k0 + kq + u;
            int ck = gk < p.K ? gk : p.K - 1;
            float v = Ab[(long long)cm * p.lda + ck];
            o[u] = (gm < p.M && gk < p.K) ? v : 0.f;
          }
        }
      }
    } else {
      if (fastAm && fk) {
#pragma unroll
        for (int i = 0; i < 2; ++i) {
          int idx = tid + i * 256;
          int k = idx >> 4, mq = (idx & 15) * 4;
          ar[i] = *(const float4*)(Ab + (long long)(k0 + k) * p.lda + (mblk + mq));
        }
      } else {
#pragma unroll
        for (int i = 0; i < 2; ++i) {
          int idx = tid + i * 256;
          int k = idx >> 4, mq = (idx & 15) * 4;
          int gk = k0 + k;
          int ck = gk < p.K ? gk : p.K - 1;
          float* o = (float*)&ar[i];
#pragma unroll
          for (int u = 0; u < 4; ++u) {
            int gm = mblk + mq + u;
            int cm = gm < p.M ? gm : p.M - 1;
            float v = Ab[(long long)ck * p.lda + cm];
            o[u] = (gm < p.M && gk < p.K) ? v : 0.f;
          }
        }
      }
    }
    if (TB == 0) {
      if (fastBn && fk) {
#pragma unroll
        for (int i = 0; i < 4; ++i) {
          int idx = tid + i * 256;
          int k = idx >> 5, nq = (idx & 31) * 4;
          br[i] = *(const float4*)(Bb + (long long)(k0 + k) * p.ldb + (nblk + nq));
        }
      } else {
#pragma unroll
        for (int i = 0; i < 4; ++i) {
          int idx = tid + i * 256;
          int k = idx >> 5, nq = (idx & 31) * 4;
          int gk = k0 + k;
          int ck = gk < p.K ? gk : p.K - 1;
          float* o = (float*)&br[i];
#pragma unroll
          for (int u = 0; u < 4; ++u) {
            int gn = nblk + nq + u;
            int cn = gn < p.N ? gn : p.N - 1;
            float v = Bb[(long long)ck * p.ldb + cn];
            o[u] = (gn < p.N && gk < p.K) ? v : 0.f;
          }
        }
      }
    } else {
      if (fastBn && fk) {
#pragma unroll
        for (int i = 0; i < 4; ++i) {
          int idx = tid + i * 256;
          int n = idx >> 3, kq = (idx & 7) * 4;
          br[i] = *(const float4*)(Bb + (long long)(nblk + n) * p.ldb + (k0 + kq));
        }
      } else {
#pragma unroll
        for (int i = 0; i < 4; ++i) {
          int idx = tid + i * 256;
          int n = idx >> 3, kq = (idx & 7) * 4;
          int gn = nblk + n;
          int cn = gn < p.N ? gn : p.N - 1;
          float* o = (float*)&br[i];
#pragma unroll
          for (int u = 0; u < 4; ++u) {
            int gk = k0 + kq + u;
            int ck = gk < p.K ? gk : p.K - 1;
            float v = Bb[(long long)cn * p.ldb + ck];
            o[u] = (gn < p.N && gk < p.K) ? v : 0.f;
          }
        }
      }
    }
  };

  // ---- convert + store staged registers into LDS buffer ----
  auto lstore = [&](int buf) {
    if (TA == 0) {
#pragma unroll
      for (int i = 0; i < 2; ++i) {
        int idx = tid + i * 256;
        int m = idx >> 3, kq = (idx & 7) * 4;
        const float* s = (const float*)&ar[i];
        union { _Float16 h[4]; unsigned long long u; } pk;
        pk.h[0] = (_Float16)s[0]; pk.h[1] = (_Float16)s[1];
        pk.h[2] = (_Float16)s[2]; pk.h[3] = (_Float16)s[3];
        *(unsigned long long*)&As[buf][m][kq] = pk.u;
      }
    } else {
#pragma unroll
      for (int i = 0; i < 2; ++i) {
        int idx = tid + i * 256;
        int k = idx >> 4, mq = (idx & 15) * 4;
        const float* s = (const float*)&ar[i];
#pragma unroll
        for (int u = 0; u < 4; ++u) As[buf][mq + u][k] = (_Float16)s[u];
      }
    }
    if (TB == 0) {
#pragma unroll
      for (int i = 0; i < 4; ++i) {
        int idx = tid + i * 256;
        int k = idx >> 5, nq = (idx & 31) * 4;
        const float* s = (const float*)&br[i];
#pragma unroll
        for (int u = 0; u < 4; ++u) Bs[buf][nq + u][k] = (_Float16)s[u];
      }
    } else {
#pragma unroll
      for (int i = 0; i < 4; ++i) {
        int idx = tid + i * 256;
        int n = idx >> 3, kq = (idx & 7) * 4;
        const float* s = (const float*)&br[i];
        union { _Float16 h[4]; unsigned long long u; } pk;
        pk.h[0] = (_Float16)s[0]; pk.h[1] = (_Float16)s[1];
        pk.h[2] = (_Float16)s[2]; pk.h[3] = (_Float16)s[3];
        *(unsigned long long*)&Bs[buf][n][kq] = pk.u;
      }
    }
  };

  v8f acc00 = {}, acc01 = {}, acc10 = {}, acc11 = {};

  const int nst = (p.K + 31) >> 5;
  gload(0);
  lstore(0);
  __syncthreads();

  for (int s = 0; s < nst; ++s) {
    const int buf = s & 1;
    const bool more = (s + 1) < nst;
    if (more) gload((s + 1) * 32);   // global loads in flight during WMMAs

    // ---- fragment loads per ISA 7.12.2 layouts + 4 WMMAs ----
    union HU { unsigned u[8]; v16h v; } a0, a1, b0, b1;
    const int koff = lhalf * 8;
#pragma unroll
    for (int q = 0; q < 8; ++q) {
      int kk = koff + ((q < 4) ? (2 * q) : (8 + 2 * q));
      a0.u[q] = *(const unsigned*)&As[buf][wm + lmod][kk];
      a1.u[q] = *(const unsigned*)&As[buf][wm + 16 + lmod][kk];
    }
    const int kb = lhalf * 8;
#pragma unroll
    for (int q = 0; q < 8; ++q) {
      int kq = kb + ((q < 4) ? (2 * q) : (8 + 2 * q));
      b0.u[q] = *(const unsigned*)&Bs[buf][wn + lmod][kq];
      b1.u[q] = *(const unsigned*)&Bs[buf][wn + 16 + lmod][kq];
    }
    acc00 = __builtin_amdgcn_wmma_f32_16x16x32_f16(false, a0.v, false, b0.v, (short)0, acc00, false, false);
    acc01 = __builtin_amdgcn_wmma_f32_16x16x32_f16(false, a0.v, false, b1.v, (short)0, acc01, false, false);
    acc10 = __builtin_amdgcn_wmma_f32_16x16x32_f16(false, a1.v, false, b0.v, (short)0, acc10, false, false);
    acc11 = __builtin_amdgcn_wmma_f32_16x16x32_f16(false, a1.v, false, b1.v, (short)0, acc11, false, false);

    if (more) lstore(buf ^ 1);       // fill the other buffer
    __syncthreads();
  }

  v8f accs[2][2] = {{acc00, acc01}, {acc10, acc11}};
#pragma unroll
  for (int r16 = 0; r16 < 2; ++r16) {
#pragma unroll
    for (int c16 = 0; c16 < 2; ++c16) {
      v8f a = accs[r16][c16];
#pragma unroll
      for (int rr = 0; rr < 8; ++rr) {
        int m = mblk + wm + r16 * 16 + lhalf * 8 + rr;   // C/D VGPR layout
        int n = nblk + wn + c16 * 16 + lmod;
        if (m < p.M && n < p.N) {
          float v = p.alpha * a[rr];
          if (p.bias) v += p.bias[n];
          if (p.act == 1) v = v > 0.f ? v : 0.f;
          else if (p.act == 2) v = 1.f / (1.f + __expf(-v));
          Cb[(long long)m * p.ldc + n] = v;
        }
      }
    }
  }
}

// ---------------------------------------------------------------------------
// Support kernels (memory-bound; VALU)
// ---------------------------------------------------------------------------
__global__ void k_init_seq(const float* t, const float* tw, const float* lmb,
                           const float* lng, const float* lnb, float* seq) {
  __shared__ float te[256];
  __shared__ float r1[512];
  __shared__ float r2[512];
  int tid = threadIdx.x;  // 512 threads
  if (tid < 128) {
    float f = __expf(-logf(10000.f) / 127.f * (float)tid);
    float a = t[0] * 10000.f * f;
    te[tid] = __sinf(a);
    te[tid + 128] = __cosf(a);
  }
  float v = (tid < 384) ? lmb[tid] : 0.f;
  r1[tid] = v; r2[tid] = v * v;
  __syncthreads();
  for (int s = 256; s > 0; s >>= 1) {
    if (tid < s) { r1[tid] += r1[tid + s]; r2[tid] += r2[tid + s]; }
    __syncthreads();
  }
  float mean = r1[0] / 384.f;
  float var  = r2[0] / 384.f - mean * mean;
  if (tid < 384) {
    float dot = 0.f;
    for (int k = 0; k < 256; ++k) dot += te[k] * tw[k * 384 + tid];
    float val = dot + (lmb[tid] - mean) * rsqrtf(var + 1e-5f) * lng[tid] + lnb[tid];
    for (int l = 0; l < 256; ++l) seq[l * 384 + tid] = val;
  }
}

__global__ void k_init_pair(const int* residx, const int* prev_pos,
                            const float* rel_emb, const float* pp_emb, float* pair) {
  int ij = blockIdx.x;              // 65536 blocks
  int i = ij >> 8, j = ij & 255;
  int c = threadIdx.x;              // 128
  int pi = residx[1 + i] - 1;
  int pj = residx[1 + j] - 1;
  int rel = (pj - pi) + 32;
  rel = rel < 0 ? 0 : (rel > 64 ? 64 : rel);
  rel += 1;
  int pp = prev_pos[ij];
  pair[(long long)ij * 128 + c] = rel_emb[rel * 128 + c] + pp_emb[pp * 128 + c];
}

__global__ void k_ln(const float* x, const float* g, const float* b, float* y, int C) {
  long long row = blockIdx.x;
  const float* xr = x + row * C;
  float* yr = y + row * C;
  __shared__ float r1[128], r2[128];
  int tid = threadIdx.x;  // 128
  float s = 0.f, s2 = 0.f;
  for (int c = tid; c < C; c += 128) { float v = xr[c]; s += v; s2 += v * v; }
  r1[tid] = s; r2[tid] = s2;
  __syncthreads();
  for (int st = 64; st > 0; st >>= 1) {
    if (tid < st) { r1[tid] += r1[tid + st]; r2[tid] += r2[tid + st]; }
    __syncthreads();
  }
  float mean = r1[0] / (float)C;
  float var  = r2[0] / (float)C - mean * mean;
  float rstd = rsqrtf(var + 1e-5f);
  for (int c = tid; c < C; c += 128) yr[c] = (xr[c] - mean) * rstd * g[c] + b[c];
}

// y[row,h] = sum_c x[row,c] * w[c,h]   (small H: attention bias projections)
__global__ void k_small_proj(const float* x, const float* w, float* y, int C, int H) {
  __shared__ float xs[384];
  long long row = blockIdx.x;
  int tid = threadIdx.x;  // 128
  for (int c = tid; c < C; c += 128) xs[c] = x[row * C + c];
  __syncthreads();
  if (tid < H) {
    float s = 0.f;
    for (int c = 0; c < C; ++c) s += xs[c] * w[c * H + tid];
    y[row * H + tid] = s;
  }
}

// softmax over 256 with pair-bias [q*Lk+k, H] and mask; in-place on logits
__global__ void k_softmax(float* lg, const float* bias, const float* mask,
                          int Lq, int H, int Lk) {
  long long row = blockIdx.x;
  int tid = threadIdx.x;  // 256
  int q = (int)(row % Lq);
  int h = (int)((row / Lq) % H);
  __shared__ float red[256];
  float v = lg[row * Lk + tid] + bias[((long long)q * Lk + tid) * H + h]
            + (mask[tid] - 1.f) * 1e9f;
  red[tid] = v;
  __syncthreads();
  for (int s = 128; s > 0; s >>= 1) {
    if (tid < s) red[tid] = fmaxf(red[tid], red[tid + s]);
    __syncthreads();
  }
  float mx = red[0];
  __syncthreads();
  float e = __expf(v - mx);
  red[tid] = e;
  __syncthreads();
  for (int s = 128; s > 0; s >>= 1) {
    if (tid < s) red[tid] += red[tid + s];
    __syncthreads();
  }
  lg[row * Lk + tid] = e / red[0];
}

// y = m * sigmoid(g) * v  (m = mask[i]*mask[j] over L x L rows if use_mask2)
__global__ void k_gate(const float* g, const float* v, float* y, const float* mask,
                       int C, int L, int use_mask2, long long total) {
  long long idx = (long long)blockIdx.x * blockDim.x + threadIdx.x;
  if (idx >= total) return;
  float m = 1.f;
  if (use_mask2) {
    long long row = idx / C;
    m = mask[(int)(row / L)] * mask[(int)(row % L)];
  }
  y[idx] = m * (1.f / (1.f + __expf(-g[idx]))) * v[idx];
}

__global__ void k_rowmask(float* x, const float* mask, int C, long long total) {
  long long idx = (long long)blockIdx.x * blockDim.x + threadIdx.x;
  if (idx >= total) return;
  x[idx] *= mask[(int)(idx / C)];
}

__global__ void k_add(float* y, const float* x, long long n) {
  long long idx = (long long)blockIdx.x * blockDim.x + threadIdx.x;
  if (idx < n) y[idx] += x[idx];
}

__global__ void k_mla(float* y, const float* a, const float* b, long long n) {
  long long idx = (long long)blockIdx.x * blockDim.x + threadIdx.x;
  if (idx < n) y[idx] += a[idx] * b[idx];
}

// y[i,j,c] += x[j,i,c]
__global__ void k_add_swap(float* y, const float* x, int L, int C) {
  long long idx = (long long)blockIdx.x * blockDim.x + threadIdx.x;
  long long total = (long long)L * L * C;
  if (idx >= total) return;
  int c = (int)(idx % C);
  int j = (int)((idx / C) % L);
  int i = (int)(idx / ((long long)C * L));
  y[idx] += x[((long long)j * L + i) * C + c];
}

// y[i,j,c] = x[j,i,c]
__global__ void k_copy_swap(float* y, const float* x, int L, int C) {
  long long idx = (long long)blockIdx.x * blockDim.x + threadIdx.x;
  long long total = (long long)L * L * C;
  if (idx >= total) return;
  int c = (int)(idx % C);
  int j = (int)((idx / C) % L);
  int i = (int)(idx / ((long long)C * L));
  y[idx] = x[((long long)j * L + i) * C + c];
}

// out[c*R + r] = in[r*Cc + c]
__global__ void k_transpose(const float* in, float* out, int R, int Cc) {
  __shared__ float t[32][33];
  int cb = blockIdx.x * 32, rb = blockIdx.y * 32;
  for (int i = threadIdx.y; i < 32; i += 8) {
    int r = rb + i, c = cb + threadIdx.x;
    if (r < R && c < Cc) t[i][threadIdx.x] = in[(long long)r * Cc + c];
  }
  __syncthreads();
  for (int i = threadIdx.y; i < 32; i += 8) {
    int ro = cb + i, co = rb + threadIdx.x;
    if (ro < Cc && co < R) out[(long long)ro * R + co] = t[threadIdx.x][i];
  }
}

// pair[i,j,n] += x[i,j,n] / (1e-3 + mask[i]*mask[j])
__global__ void k_opm_add(float* pair, const float* x, const float* mask, int L, int C) {
  long long idx = (long long)blockIdx.x * blockDim.x + threadIdx.x;
  long long total = (long long)L * L * C;
  if (idx >= total) return;
  int j = (int)((idx / C) % L);
  int i = (int)(idx / ((long long)C * L));
  pair[idx] += x[idx] / (1e-3f + mask[i] * mask[j]);
}

// ---------------------------------------------------------------------------
// Host-side parameter tree (jax pytree leaf order: sorted dict keys, DFS)
// ---------------------------------------------------------------------------
struct HLin { const float* w; const float* b; };
struct HLN  { const float* g; const float* b; };
struct HAttn { HLin bias, gate, k, out, q, v; HLN ln, ln_pair; };
struct HTri  { HLin bias, gate, k, out, q, v; HLN ln; };
struct HTM   { HLin gate, lg, lp, out, rg, rp; HLN ln, ln_out; };
struct HTrans{ HLin l1, l2; HLN ln; };
struct HOPM  { HLin left, right, out; HLN ln; };
struct HBlock{ HOPM opm; HTrans pt; HAttn sa; HTrans st; HTri tae, tas; HTM tmi, tmo; };

extern "C" void kernel_launch(void* const* d_in, const int* in_sizes, int n_in,
                              void* d_out, int out_size, void* d_ws, size_t ws_size,
                              hipStream_t stream) {
  (void)in_sizes; (void)n_in; (void)out_size; (void)ws_size;
  const int L = 256;

  const int*   residx   = (const int*)d_in[2];
  const float* maskp    = (const float*)d_in[1];
  const float* tparam   = (const float*)d_in[3];
  const int*   prev_pos = (const int*)d_in[4];

  // ---- walk params pytree in sorted-key leaf order ----
  int ci = 5;
  auto nx = [&]() -> const float* { return (const float*)d_in[ci++]; };
  auto rdLin = [&](HLin& l, bool hasB) { l.b = hasB ? nx() : nullptr; l.w = nx(); };
  auto rdLN  = [&](HLN& n) { n.b = nx(); n.g = nx(); };

  HBlock B[2];
  for (int bi = 0; bi < 2; ++bi) {
    HBlock& b = B[bi];
    // opm: left, ln, out, right
    rdLin(b.opm.left, true); rdLN(b.opm.ln); rdLin(b.opm.out, true); rdLin(b.opm.right, true);
    // pair_trans: l1, l2, ln
    rdLin(b.pt.l1, true); rdLin(b.pt.l2, true); rdLN(b.pt.ln);
    // seq_attn: bias, gate, k, ln, ln_pair, out, q, v
    rdLin(b.sa.bias, false); rdLin(b.sa.gate, true); rdLin(b.sa.k, false);
    rdLN(b.sa.ln); rdLN(b.sa.ln_pair); rdLin(b.sa.out, true);
    rdLin(b.sa.q, false); rdLin(b.sa.v, false);
    // seq_trans: l1, l2, ln
    rdLin(b.st.l1, true); rdLin(b.st.l2, true); rdLN(b.st.ln);
    // ta_end then ta_start: bias, gate, k, ln, out, q, v
    auto rdTri = [&](HTri& t) {
      rdLin(t.bias, false); rdLin(t.gate, true); rdLin(t.k, false);
      rdLN(t.ln); rdLin(t.out, true); rdLin(t.q, false); rdLin(t.v, false);
    };
    rdTri(b.tae); rdTri(b.tas);
    // tm_in then tm_out: gate, lg, ln, ln_out, lp, out, rg, rp
    auto rdTM = [&](HTM& t) {
      rdLin(t.gate, true); rdLin(t.lg, true); rdLN(t.ln); rdLN(t.ln_out);
      rdLin(t.lp, true); rdLin(t.out, true); rdLin(t.rg, true); rdLin(t.rp, true);
    };
    rdTM(b.tmi); rdTM(b.tmo);
  }
  HLN lm_norm;  rdLN(lm_norm);
  HLin lm_proj; rdLin(lm_proj, true);
  const float* pp_emb  = nx();
  const float* rel_emb = nx();
  HLin t_proj; rdLin(t_proj, false);

  // ---- workspace layout (floats) ----
  float* W   = (float*)d_ws;
  float* LOG = W;                     // 67,108,864  (tri logits [i,h,q,k])
  float* ZLN = LOG + 67108864LL;      // 8,388,608
  float* Qb  = ZLN + 8388608LL;
  float* Kb  = Qb  + 8388608LL;
  float* Vb  = Kb  + 8388608LL;
  float* Gb  = Vb  + 8388608LL;
  float* OB  = Gb  + 8388608LL;
  float* H1  = OB  + 8388608LL;       // 33,554,432 (ffn hidden / LT)
  float* T1  = H1  + 33554432LL;
  float* T2  = T1  + 8388608LL;
  float* BIA = T2  + 8388608LL;       // 524,288
  float* SLN = BIA + 524288LL;        // 98,304

  float* seqO  = (float*)d_out;            // [256,384]
  float* pairO = seqO + 98304LL;           // [256,256,128]

  // ---- launch helpers ----
  auto gemmL = [&](GP p, int batch, int tA, int tB) {
    dim3 grid((p.N + 127) / 128, (p.M + 63) / 64, batch);
    if (tA == 0 && tB == 0)      k_gemm<0, 0><<<grid, dim3(256), 0, stream>>>(p);
    else if (tA == 0 && tB == 1) k_gemm<0, 1><<<grid, dim3(256), 0, stream>>>(p);
    else if (tA == 1 && tB == 0) k_gemm<1, 0><<<grid, dim3(256), 0, stream>>>(p);
    else                         k_gemm<1, 1><<<grid, dim3(256), 0, stream>>>(p);
  };
  auto mk = [](const float* A, const float* Bm, const float* bias, float* C,
               int M, int N, int K, int lda, int ldb, int ldc,
               float alpha, int act) {
    GP p; p.A = A; p.B = Bm; p.bias = bias; p.C = C;
    p.M = M; p.N = N; p.K = K; p.lda = lda; p.ldb = ldb; p.ldc = ldc;
    p.act = act; p.batch_inner = 1; p.alpha = alpha;
    p.sAi = p.sAo = p.sBi = p.sBo = p.sCi = p.sCo = 0;
    return p;
  };
  auto lin = [&](const float* x, HLin wv, float* y, int M, int K, int N,
                 int act, bool useBias) {
    gemmL(mk(x, wv.w, useBias ? wv.b : nullptr, y, M, N, K, K, N, N, 1.f, act),
          1, 0, 0);
  };
  auto ln = [&](const float* x, HLN p, float* y, int rows, int C) {
    k_ln<<<dim3(rows), dim3(128), 0, stream>>>(x, p.g, p.b, y, C);
  };
  auto ewg = [](long long n) { return dim3((unsigned)((n + 255) / 256)); };
  auto add = [&](float* y, const float* x, long long n) {
    k_add<<<ewg(n), dim3(256), 0, stream>>>(y, x, n);
  };

  // ---- embeddings ----
  k_init_seq<<<dim3(1), dim3(512), 0, stream>>>(tparam, t_proj.w, lm_proj.b,
                                                lm_norm.g, lm_norm.b, seqO);
  k_init_pair<<<dim3(65536), dim3(128), 0, stream>>>(residx, prev_pos, rel_emb,
                                                     pp_emb, pairO);

  // ---- module implementations ----
  auto seq_attn = [&](const HAttn& P) {
    ln(seqO, P.ln, SLN, 256, 384);
    ln(pairO, P.ln_pair, ZLN, 65536, 128);
    k_small_proj<<<dim3(65536), dim3(128), 0, stream>>>(ZLN, P.bias.w, BIA, 128, 8);
    lin(SLN, P.q, Qb, 256, 384, 384, 0, false);
    lin(SLN, P.k, Kb, 256, 384, 384, 0, false);
    lin(SLN, P.v, Vb, 256, 384, 384, 0, false);
    lin(SLN, P.gate, Gb, 256, 384, 384, 0, true);
    GP lg = mk(Qb, Kb, nullptr, LOG, 256, 256, 48, 384, 384, 256,
               1.f / sqrtf(48.f), 0);
    lg.batch_inner = 8; lg.sAi = 48; lg.sBi = 48; lg.sCi = 65536;
    gemmL(lg, 8, 0, 1);
    k_softmax<<<dim3(2048), dim3(256), 0, stream>>>(LOG, BIA, maskp, 256, 8, 256);
    GP av = mk(LOG, Vb, nullptr, OB, 256, 48, 256, 256, 384, 384, 1.f, 0);
    av.batch_inner = 8; av.sAi = 65536; av.sBi = 48; av.sCi = 48;
    gemmL(av, 8, 0, 0);
    k_gate<<<ewg(98304), dim3(256), 0, stream>>>(Gb, OB, OB, maskp, 384, L, 0, 98304);
    lin(OB, P.out, T1, 256, 384, 384, 0, true);
    add(seqO, T1, 98304);
  };

  auto transition = [&](float* x, const HTrans& P, int rows, int C, float* lnbuf) {
    ln(x, P.ln, lnbuf, rows, C);
    lin(lnbuf, P.l1, H1, rows, C, 4 * C, 1, true);
    lin(H1, P.l2, T1, rows, 4 * C, C, 0, true);
    add(x, T1, (long long)rows * C);
  };

  auto opm = [&](const HOPM& P) {
    ln(seqO, P.ln, SLN, 256, 384);
    lin(SLN, P.left, Qb, 256, 384, 32, 0, true);
    lin(SLN, P.right, Kb, 256, 384, 32, 0, true);
    k_rowmask<<<ewg(8192), dim3(256), 0, stream>>>(Qb, maskp, 32, 8192);
    k_rowmask<<<ewg(8192), dim3(256), 0, stream>>>(Kb, maskp, 32, 8192);
    // stage1: Ttmp[a][j,n] = sum_c right[j,c] * W[a*32+c, n]
    GP s1 = mk(Kb, P.out.w, nullptr, T1, 256, 128, 32, 32, 128, 128, 1.f, 0);
    s1.batch_inner = 32; s1.sBi = 4096; s1.sCi = 32768;
    gemmL(s1, 32, 0, 0);
    // stage2: out[i,j,n] = sum_a left[i,a] * Ttmp[a][j,n]
    GP s2 = mk(Qb, T1, P.out.b, T2, 256, 128, 32, 32, 32768, 32768, 1.f, 0);
    s2.batch_inner = 256; s2.sBi = 128; s2.sCi = 128;
    gemmL(s2, 256, 0, 0);
    k_opm_add<<<ewg(8388608), dim3(256), 0, stream>>>(pairO, T2, maskp, L, 128);
  };

  auto tri_mult = [&](const HTM& P, bool outgoing) {
    ln(pairO, P.ln, ZLN, 65536, 128);
    lin(ZLN, P.lp, Qb, 65536, 128, 128, 0, true);
    lin(ZLN, P.lg, Kb, 65536, 128, 128, 0, true);
    lin(ZLN, P.rp, Vb, 65536, 128, 128, 0, true);
    lin(ZLN, P.rg, Gb, 65536, 128, 128, 0, true);
    k_gate<<<ewg(8388608), dim3(256), 0, stream>>>(Kb, Qb, Qb, maskp, 128, L, 1, 8388608);
    k_gate<<<ewg(8388608), dim3(256), 0, stream>>>(Gb, Vb, Vb, maskp, 128, L, 1, 8388608);
    // channel-major for coalesced batched GEMMs
    k_transpose<<<dim3(4, 2048), dim3(32, 8), 0, stream>>>(Qb, H1, 65536, 128);  // LT
    k_transpose<<<dim3(4, 2048), dim3(32, 8), 0, stream>>>(Vb, T1, 65536, 128);  // RT
    // outgoing: o_c[i,j] = sum_k LT[c][i,k]*RT[c][j,k]  (N x T)
    // incoming: o_c[i,j] = sum_k LT[c][k,i]*RT[c][k,j]  (T x N)
    GP pe = mk(H1, T1, nullptr, T2, 256, 256, 256, 256, 256, 256, 1.f, 0);
    pe.batch_inner = 128; pe.sAi = 65536; pe.sBi = 65536; pe.sCi = 65536;
    gemmL(pe, 128, outgoing ? 0 : 1, outgoing ? 1 : 0);
    k_transpose<<<dim3(2048, 4), dim3(32, 8), 0, stream>>>(T2, OB, 128, 65536);  // o [ij,c]
    ln(OB, P.ln_out, Qb, 65536, 128);
    lin(ZLN, P.gate, Gb, 65536, 128, 128, 2, true);    // sigmoid gate
    lin(Qb, P.out, Kb, 65536, 128, 128, 0, true);
    k_mla<<<ewg(8388608), dim3(256), 0, stream>>>(pairO, Gb, Kb, 8388608);
  };

  auto tri_attn = [&](const HTri& P, bool starting) {
    const float* zin = pairO;
    if (!starting) {
      k_copy_swap<<<ewg(8388608), dim3(256), 0, stream>>>(T2, pairO, L, 128);
      zin = T2;
    }
    ln(zin, P.ln, ZLN, 65536, 128);
    k_small_proj<<<dim3(65536), dim3(128), 0, stream>>>(ZLN, P.bias.w, BIA, 128, 4);
    lin(ZLN, P.q, Qb, 65536, 128, 128, 0, false);
    lin(ZLN, P.k, Kb, 65536, 128, 128, 0, false);
    lin(ZLN, P.v, Vb, 65536, 128, 128, 0, false);
    lin(ZLN, P.gate, Gb, 65536, 128, 128, 0, true);
    GP lg = mk(Qb, Kb, nullptr, LOG, 256, 256, 32, 128, 128, 256,
               1.f / sqrtf(32.f), 0);
    lg.batch_inner = 4;
    lg.sAi = 32; lg.sAo = 32768; lg.sBi = 32; lg.sBo = 32768;
    lg.sCi = 65536; lg.sCo = 262144;
    gemmL(lg, 1024, 0, 1);
    k_softmax<<<dim3(262144), dim3(256), 0, stream>>>(LOG, BIA, maskp, 256, 4, 256);
    GP av = mk(LOG, Vb, nullptr, OB, 256, 32, 256, 256, 128, 128, 1.f, 0);
    av.batch_inner = 4;
    av.sAi = 65536; av.sAo = 262144; av.sBi = 32; av.sBo = 32768;
    av.sCi = 32; av.sCo = 32768;
    gemmL(av, 1024, 0, 0);
    k_gate<<<ewg(8388608), dim3(256), 0, stream>>>(Gb, OB, OB, maskp, 128, L, 0, 8388608);
    lin(OB, P.out, T1, 65536, 128, 128, 0, true);
    if (starting) add(pairO, T1, 8388608);
    else k_add_swap<<<ewg(8388608), dim3(256), 0, stream>>>(pairO, T1, L, 128);
  };

  // ---- seqformer blocks ----
  for (int bi = 0; bi < 2; ++bi) {
    HBlock& b = B[bi];
    seq_attn(b.sa);
    transition(seqO, b.st, 256, 384, SLN);
    opm(b.opm);
    tri_mult(b.tmo, true);
    tri_mult(b.tmi, false);
    tri_attn(b.tas, true);
    tri_attn(b.tae, false);
    transition(pairO, b.pt, 65536, 128, ZLN);
  }
}
